// Att_Prior_70600672412062
// MI455X (gfx1250) — matrix operand order repaired and checked by reading, not compiled, not run on hardware
//
#include <hip/hip_runtime.h>
#include <math.h>
#include <stdint.h>

#define B_  16
#define C_  8
#define L_  1024
#define NH_ 5
#define HD_ 80
#define DK_ 400
#define M_  (B_ * L_)   // 16384 rows

typedef _Float16 half_t;
typedef __attribute__((ext_vector_type(16))) _Float16 v16h;
typedef __attribute__((ext_vector_type(8)))  _Float16 v8h;
typedef __attribute__((ext_vector_type(8)))  float    v8f;
typedef __attribute__((ext_vector_type(4)))  int      i32x4;

static __device__ __forceinline__ v16h cat8(v8h lo, v8h hi) {
  v16h r;
#pragma unroll
  for (int i = 0; i < 8; ++i) { r[i] = lo[i]; r[i + 8] = hi[i]; }
  return r;
}

// ---- CDNA5 async global->LDS copy (guarded; falls back to sync copy) -------
// Probe result (round 2 diagnostic): builtin exists; params are
// (int4 AS1* src, int4 AS3* dst, imm int offset, imm int cpol).
#if defined(__has_builtin)
#if __has_builtin(__builtin_amdgcn_global_load_async_to_lds_b128) && \
    __has_builtin(__builtin_amdgcn_s_wait_asynccnt)
#define USE_ASYNC_LDS 1
#endif
#endif

typedef i32x4 __attribute__((address_space(1)))* as1_v4i;
typedef i32x4 __attribute__((address_space(3)))* as3_v4i;

static __device__ __forceinline__ void copy16_g2l(const half_t* g, half_t* l) {
#ifdef USE_ASYNC_LDS
  __builtin_amdgcn_global_load_async_to_lds_b128(
      (as1_v4i)(uintptr_t)g, (as3_v4i)(uint32_t)(uintptr_t)l, 0, 0);
#else
  *(v8h*)l = *(const v8h*)g;
#endif
}
static __device__ __forceinline__ void wait_async() {
#ifdef USE_ASYNC_LDS
  __builtin_amdgcn_s_wait_asynccnt(0);
#endif
}

// ---------------------------------------------------------------------------
// Weight pack: f32 [K,N] -> f16 WMMA B-panels, zero padded.
// Layout: dst[((nt*KC + kc)*32 + lane)*16 + i] = W[kc*32 + (lane>>4)*16 + i][nt*16 + (lane&15)]
// so a lane's B-fragment for (nt,kc) is one contiguous 32B run.
// ---------------------------------------------------------------------------
__global__ __launch_bounds__(256)
void pack_wpanel_kernel(const float* __restrict__ src, half_t* __restrict__ dst,
                        int K, int N, int Kpad, int Npad)
{
  const int KC = Kpad >> 5;
  const int total = (Npad >> 4) * KC * 512;
  for (int idx = blockIdx.x * 256 + threadIdx.x; idx < total; idx += gridDim.x * 256) {
    const int i    = idx & 15;
    const int lane = (idx >> 4) & 31;
    const int kc   = (idx >> 9) % KC;
    const int nt   = idx / (KC * 512);
    const int k = kc * 32 + (lane >> 4) * 16 + i;
    const int n = nt * 16 + (lane & 15);
    float v = 0.f;
    if (k < K && n < N) v = src[(size_t)k * N + n];
    dst[idx] = (half_t)v;
  }
}

// ---------------------------------------------------------------------------
// Build z: shifted input channels + positional encodings, f16 [M,32] (K=11 pad)
// ---------------------------------------------------------------------------
__global__ __launch_bounds__(256)
void build_z_kernel(const float* __restrict__ x, half_t* __restrict__ z)
{
  const int idx = blockIdx.x * 256 + threadIdx.x;
  if (idx >= M_ * 32) return;
  const int colz = idx & 31;
  const int m = idx >> 5;
  const int b = m >> 10, l = m & (L_ - 1);
  float v = 0.f;
  if (colz < C_) {
    if (l > 0) v = x[((size_t)(b * C_ + colz)) * L_ + (l - 1)];
  } else if (colz == C_) {
    v = (float)l / 1023.f;
  } else if (colz == C_ + 1) {
    v = 10.f * (float)l / 1023.f;
  } else if (colz == C_ + 2) {
    v = -1.f + 2.f * (float)l / 1023.f;
  }
  z[idx] = (half_t)v;
}

// ---------------------------------------------------------------------------
// WMMA GEMM, LDS-free (all operands L2-resident on MI455X's 192MB L2):
// out = act(A_f16[M,Kpad] @ W + bias). Each wave: 64 rows x 16 cols
// (4 accumulators, 4 WMMAs per B-fragment load). Block = 8 waves = 512 rows.
// B-fragments come pre-packed in per-lane panel layout (2 coalesced b128/chunk).
// ---------------------------------------------------------------------------
__global__ __launch_bounds__(256)
void gemm_wmma_kernel(const half_t* __restrict__ A, int lda,
                      const half_t* __restrict__ Wp, int KC,
                      const float* __restrict__ bias, int Nreal,
                      float* out32, int ldo32, int accum,
                      half_t* out16, int ldo16, int N16, int act)
{
  const int lane  = threadIdx.x & 31;
  const int wave  = threadIdx.x >> 5;
  const int group = lane >> 4;
  const int col   = lane & 15;
  const int nt = blockIdx.x;
  const int m0 = blockIdx.y * 512 + wave * 64;

  const half_t* wp = Wp + (size_t)nt * KC * 512;
  v8f acc[4];
#pragma unroll
  for (int t = 0; t < 4; ++t) { v8f z = {}; acc[t] = z; }

  for (int kc = 0; kc < KC; ++kc) {
    const half_t* bp = wp + ((size_t)kc * 32 + lane) * 16;
    if (kc + 1 < KC)
      __builtin_prefetch(wp + ((size_t)(kc + 1) * 32 + lane) * 16, 0, 1);
    v16h bf = cat8(*(const v8h*)bp, *(const v8h*)(bp + 8));
    const int kbase = kc * 32;
#pragma unroll
    for (int t = 0; t < 4; ++t) {
      const half_t* ap = &A[(size_t)(m0 + t * 16 + col) * lda + kbase];
      v8h alo = *(const v8h*)(ap + group * 16);
      v8h ahi = *(const v8h*)(ap + group * 16 + 8);
      acc[t] = __builtin_amdgcn_wmma_f32_16x16x32_f16(false, cat8(alo, ahi),
                                                      false, bf,
                                                      (short)0, acc[t], false, false);
    }
  }

  const int colg = nt * 16 + col;
  const float bv = (bias != nullptr && colg < Nreal) ? bias[colg] : 0.f;
#pragma unroll
  for (int t = 0; t < 4; ++t) {
#pragma unroll
    for (int r = 0; r < 8; ++r) {
      const int row = m0 + t * 16 + r + 8 * group;   // C layout: M = r + 8*group
      float vv = acc[t][r] + bv;
      if (act == 1) vv = vv > 0.f ? vv : 0.01f * vv; // leaky_relu(0.01)
      if (out32 != nullptr && colg < Nreal) {
        float* p = &out32[(size_t)row * ldo32 + colg];
        *p = accum ? (*p + vv) : vv;
      }
      if (out16 != nullptr && colg < N16)
        out16[(size_t)row * ldo16 + colg] = (half_t)vv;
    }
  }
}

// ---------------------------------------------------------------------------
// Flash-style causal attention: one block per (b,h, 128 q-rows), 8 waves.
// q/k/v f16 [M,400] head-major cols; O f16 [M,416] (padded Kpad for m1 GEMM).
// Q/K/V tiles staged into LDS with async b128 copies; V transposed in LDS.
// ---------------------------------------------------------------------------
__global__ __launch_bounds__(256)
void attn_kernel(const half_t* __restrict__ q, const half_t* __restrict__ k,
                 const half_t* __restrict__ v, half_t* __restrict__ o)
{
  __shared__ __align__(16) half_t Qs[128 * 96];    // q rows, d padded 80->96
  __shared__ __align__(16) half_t Ks[32 * 96];     // key-major (B layout for QK^T)
  __shared__ __align__(16) half_t Vs[32 * 80];     // key-major staging for V
  __shared__ __align__(16) half_t VT[80 * 32];     // d-major (B layout for AV)
  __shared__ __align__(16) half_t Ps[8 * 16 * 32]; // per-wave P scratch

  const int tid   = threadIdx.x;
  const int lane  = tid & 31;
  const int wave  = tid >> 5;
  const int group = lane >> 4;
  const int col   = lane & 15;
  const int bh = blockIdx.x;
  const int b = bh / NH_, h = bh % NH_;
  const int qb = blockIdx.y;
  const int q0 = qb * 128 + wave * 16;

  // d-padding (80..95) of Qs/Ks is constant: write zeros once
  for (int i = tid; i < 128 * 16; i += 256) { int r = i >> 4; Qs[r * 96 + 80 + (i & 15)] = (half_t)0; }
  for (int i = tid; i < 32 * 16; i += 256)  { int r = i >> 4; Ks[r * 96 + 80 + (i & 15)] = (half_t)0; }
  // stage Q tile (10 x 16B chunks per row)
  for (int i = tid; i < 128 * 10; i += 256) {
    int r = i / 10, cc = i % 10;
    copy16_g2l(&q[((size_t)(b * L_ + qb * 128 + r)) * DK_ + h * HD_ + cc * 8],
               &Qs[r * 96 + cc * 8]);
  }
  wait_async();
  __syncthreads();

  v8f zero = {};
  v8f acc[5];
#pragma unroll
  for (int dt = 0; dt < 5; ++dt) acc[dt] = zero;
  float m_r[8], l_r[8];
#pragma unroll
  for (int r = 0; r < 8; ++r) { m_r[r] = -INFINITY; l_r[r] = 0.f; }

  const float scale = 0.11180339887498949f;  // 1/sqrt(80)
  const int nchunks = (qb + 1) * 4;          // causal: keys <= qb*128+127
  half_t* Pw = &Ps[wave * 16 * 32];

  for (int kc = 0; kc < nchunks; ++kc) {
    __syncthreads();
    const int key0 = kc * 32;
    for (int i = tid; i < 32 * 10; i += 256) {
      int r = i / 10, cc = i % 10;
      copy16_g2l(&k[((size_t)(b * L_ + key0 + r)) * DK_ + h * HD_ + cc * 8],
                 &Ks[r * 96 + cc * 8]);
    }
    for (int i = tid; i < 32 * 10; i += 256) {
      int r = i / 10, cc = i % 10;
      copy16_g2l(&v[((size_t)(b * L_ + key0 + r)) * DK_ + h * HD_ + cc * 8],
                 &Vs[r * 80 + cc * 8]);
    }
    wait_async();
    __syncthreads();
    // transpose V in LDS: VT[d][key]
    for (int i = tid; i < 80 * 32; i += 256) {
      int d = i / 32, kk = i & 31;
      VT[i] = Vs[kk * 80 + d];
    }
    __syncthreads();

    if (key0 > q0 + 15) continue;   // fully masked for this wave's q tile

#pragma unroll
    for (int nt = 0; nt < 2; ++nt) {
      v8f s = {};
#pragma unroll
      for (int dc = 0; dc < 3; ++dc) {     // d = 0..95 (padded)
        v8h alo = *(const v8h*)&Qs[(wave * 16 + col) * 96 + dc * 32 + group * 16];
        v8h ahi = *(const v8h*)&Qs[(wave * 16 + col) * 96 + dc * 32 + group * 16 + 8];
        v8h blo = *(const v8h*)&Ks[(nt * 16 + col) * 96 + dc * 32 + group * 16];
        v8h bhi = *(const v8h*)&Ks[(nt * 16 + col) * 96 + dc * 32 + group * 16 + 8];
        s = __builtin_amdgcn_wmma_f32_16x16x32_f16(false, cat8(alo, ahi),
                                                   false, cat8(blo, bhi),
                                                   (short)0, s, false, false);
      }
      const int kg = key0 + nt * 16 + col;
#pragma unroll
      for (int r = 0; r < 8; ++r) {
        const int qg = q0 + r + 8 * group;
        float sv = (kg <= qg) ? (float)s[r] * scale : -INFINITY;
        float mx = sv;
#pragma unroll
        for (int off = 1; off < 16; off <<= 1)
          mx = fmaxf(mx, __shfl_xor(mx, off, 32));
        const float m_new = fmaxf(m_r[r], mx);
        float p = __expf(sv - m_new);
        float ps = p;
#pragma unroll
        for (int off = 1; off < 16; off <<= 1)
          ps += __shfl_xor(ps, off, 32);
        const float al = __expf(m_r[r] - m_new);
        l_r[r] = l_r[r] * al + ps;
        m_r[r] = m_new;
#pragma unroll
        for (int dt = 0; dt < 5; ++dt) acc[dt][r] *= al;
        Pw[(r + 8 * group) * 32 + nt * 16 + col] = (half_t)p;
      }
    }
    // O += P(16x32) @ V(32x16) for each of 5 d-tiles
    {
      v8h alo = *(const v8h*)&Pw[col * 32 + group * 16];
      v8h ahi = *(const v8h*)&Pw[col * 32 + group * 16 + 8];
      v16h a = cat8(alo, ahi);
#pragma unroll
      for (int dt = 0; dt < 5; ++dt) {
        v8h blo = *(const v8h*)&VT[(dt * 16 + col) * 32 + group * 16];
        v8h bhi = *(const v8h*)&VT[(dt * 16 + col) * 32 + group * 16 + 8];
        acc[dt] = __builtin_amdgcn_wmma_f32_16x16x32_f16(false, a,
                                                         false, cat8(blo, bhi),
                                                         (short)0, acc[dt], false, false);
      }
    }
  }

#pragma unroll
  for (int r = 0; r < 8; ++r) {
    const float invl = 1.f / l_r[r];
    const size_t row = (size_t)(b * L_ + q0 + r + 8 * group);
#pragma unroll
    for (int dt = 0; dt < 5; ++dt)
      o[row * 416 + h * HD_ + dt * 16 + col] = (half_t)(acc[dt][r] * invl);
  }
}

// ---------------------------------------------------------------------------
// Row norm (torch-style unbiased std + 1e-6), f16 padded output (pads = 0)
// ---------------------------------------------------------------------------
__global__ __launch_bounds__(256)
void norm_kernel(const float* __restrict__ hsrc, const float* __restrict__ alpha,
                 const float* __restrict__ bias, half_t* __restrict__ out,
                 int N, int ldh, int ldo)
{
  __shared__ float red[256];
  const int row = blockIdx.x;
  const float* x = hsrc + (size_t)row * ldh;
  float s = 0.f;
  for (int i = threadIdx.x; i < N; i += 256) s += x[i];
  red[threadIdx.x] = s; __syncthreads();
  for (int off = 128; off > 0; off >>= 1) {
    if (threadIdx.x < off) red[threadIdx.x] += red[threadIdx.x + off];
    __syncthreads();
  }
  const float mu = red[0] / (float)N;
  __syncthreads();
  float v = 0.f;
  for (int i = threadIdx.x; i < N; i += 256) { float d = x[i] - mu; v += d * d; }
  red[threadIdx.x] = v; __syncthreads();
  for (int off = 128; off > 0; off >>= 1) {
    if (threadIdx.x < off) red[threadIdx.x] += red[threadIdx.x + off];
    __syncthreads();
  }
  const float inv = 1.f / (sqrtf(red[0] / (float)(N - 1)) + 1e-6f);
  for (int i = threadIdx.x; i < ldo; i += 256) {
    float o = 0.f;
    if (i < N) o = alpha[i] * (x[i] - mu) * inv + bias[i];
    out[(size_t)row * ldo + i] = (half_t)o;
  }
}

// ---------------------------------------------------------------------------
// Log-likelihood reduction: one block per batch element
// ---------------------------------------------------------------------------
__global__ __launch_bounds__(256)
void ll_kernel(const float* __restrict__ hh, const float* __restrict__ x,
               const float* __restrict__ objective, float* __restrict__ outObj)
{
  __shared__ float red[256];
  const int b = blockIdx.x;
  const float LOG2PI = 1.8378770664093453f;
  float s = 0.f;
  for (int i = threadIdx.x; i < C_ * L_; i += 256) {
    const int c = i >> 10, l = i & (L_ - 1);
    const float mraw = hh[((size_t)(b * L_ + l)) * 16 + c];
    const float graw = hh[((size_t)(b * L_ + l)) * 16 + c + 8];
    const float mean  = tanhf(mraw * 0.01f) * 100.f;
    const float logsd = tanhf(graw * 0.1f) * 10.f - 5.f;
    const float xv = x[((size_t)(b * C_ + c)) * L_ + l];
    const float d = xv - mean;
    s += -0.5f * (LOG2PI + 2.f * logsd + d * d * __expf(-2.f * logsd));
  }
  red[threadIdx.x] = s; __syncthreads();
  for (int off = 128; off > 0; off >>= 1) {
    if (threadIdx.x < off) red[threadIdx.x] += red[threadIdx.x + off];
    __syncthreads();
  }
  if (threadIdx.x == 0) outObj[b] = objective[b] + red[0];
}

// ---------------------------------------------------------------------------
extern "C" void kernel_launch(void* const* d_in, const int* in_sizes, int n_in,
                              void* d_out, int out_size, void* d_ws, size_t ws_size,
                              hipStream_t stream)
{
  (void)in_sizes; (void)n_in; (void)out_size; (void)ws_size;
  const float* x         = (const float*)d_in[0];
  const float* objective = (const float*)d_in[1];
  const float* att1[10]; const float* att2[10]; const float* att3[10];
  for (int i = 0; i < 10; ++i) {
    att1[i] = (const float*)d_in[2 + i];
    att2[i] = (const float*)d_in[12 + i];
    att3[i] = (const float*)d_in[22 + i];
  }
  const float* na[3] = { (const float*)d_in[32], (const float*)d_in[34], (const float*)d_in[36] };
  const float* nb[3] = { (const float*)d_in[33], (const float*)d_in[35], (const float*)d_in[37] };
  const float* f1w = (const float*)d_in[38]; const float* f1b = (const float*)d_in[39];
  const float* f2w = (const float*)d_in[40]; const float* f2b = (const float*)d_in[41];
  const float** attp[3] = { att1, att2, att3 };

  size_t off = 0;
  auto alloc = [&](size_t bytes) -> void* {
    void* p = (void*)((char*)d_ws + off);
    off += (bytes + 255) & ~(size_t)255;
    return p;
  };
  half_t* z16  = (half_t*)alloc((size_t)M_ * 32 * 2);
  half_t* q16  = (half_t*)alloc((size_t)M_ * DK_ * 2);
  half_t* k16  = (half_t*)alloc((size_t)M_ * DK_ * 2);
  half_t* v16  = (half_t*)alloc((size_t)M_ * DK_ * 2);
  half_t* g16  = q16;  // reuse qkv region for final-MLP hidden (33.5MB <= 39.3MB)
  half_t* o16  = (half_t*)alloc((size_t)M_ * 416 * 2);
  half_t* h116 = (half_t*)alloc((size_t)M_ * 32 * 2);
  float*  hF   = (float*)alloc((size_t)M_ * 1000 * 4);
  half_t* n16  = (half_t*)alloc((size_t)M_ * 1024 * 2);
  float*  hh   = (float*)alloc((size_t)M_ * 16 * 4);

  const int kin[3]    = { 11, 1000, 1000 };
  const int kinPad[3] = { 32, 1024, 1024 };
  half_t *wq[3], *wk[3], *wv[3], *wm1[3], *wm2[3];
  for (int t = 0; t < 3; ++t) {
    wq[t]  = (half_t*)alloc((size_t)kinPad[t] * DK_ * 2);
    wk[t]  = (half_t*)alloc((size_t)kinPad[t] * DK_ * 2);
    wv[t]  = (half_t*)alloc((size_t)kinPad[t] * DK_ * 2);
    wm1[t] = (half_t*)alloc((size_t)416 * 32 * 2);
    wm2[t] = (half_t*)alloc((size_t)32 * 1008 * 2);
  }
  half_t* wf1 = (half_t*)alloc((size_t)1024 * 1008 * 2);
  half_t* wf2 = (half_t*)alloc((size_t)1024 * 16 * 2);

  auto pack = [&](const float* src, half_t* dst, int K, int N, int Kpad, int Npad) {
    int total = Kpad * Npad;
    int blocks = (total + 255) / 256; if (blocks > 4096) blocks = 4096;
    pack_wpanel_kernel<<<blocks, 256, 0, stream>>>(src, dst, K, N, Kpad, Npad);
  };
  for (int t = 0; t < 3; ++t) {
    pack(attp[t][0], wq[t],  kin[t], DK_, kinPad[t], DK_);
    pack(attp[t][2], wk[t],  kin[t], DK_, kinPad[t], DK_);
    pack(attp[t][4], wv[t],  kin[t], DK_, kinPad[t], DK_);
    pack(attp[t][6], wm1[t], DK_, 20, 416, 32);
    pack(attp[t][8], wm2[t], 20, 1000, 32, 1008);
  }
  pack(f1w, wf1, 1000, 1000, 1024, 1008);
  pack(f2w, wf2, 1000, 16, 1024, 16);

  build_z_kernel<<<(M_ * 32) / 256, 256, 0, stream>>>(x, z16);
  (void)hipMemsetAsync(o16, 0, (size_t)M_ * 416 * 2, stream); // zero Kpad cols 400..415

  // gemm(A, lda(=Kpad), Wpanels, Npad, bias, Nreal, out32, ldo32, accum, out16, ldo16, N16, act)
  auto gemm = [&](const half_t* A, int lda, const half_t* W, int Npad,
                  const float* bias, int Nreal,
                  float* out32, int ldo32, int accum,
                  half_t* out16, int ldo16, int N16, int act) {
    dim3 grid(Npad / 16, M_ / 512);
    gemm_wmma_kernel<<<grid, 256, 0, stream>>>(A, lda, W, lda / 32, bias, Nreal,
                                               out32, ldo32, accum, out16, ldo16, N16, act);
  };

  const half_t* Ablk = z16;
  int ldA = 32;
  for (int t = 0; t < 3; ++t) {
    gemm(Ablk, ldA, wq[t], DK_, attp[t][1], DK_, nullptr, 0, 0, q16, DK_, DK_, 0);
    gemm(Ablk, ldA, wk[t], DK_, attp[t][3], DK_, nullptr, 0, 0, k16, DK_, DK_, 0);
    gemm(Ablk, ldA, wv[t], DK_, attp[t][5], DK_, nullptr, 0, 0, v16, DK_, DK_, 0);
    attn_kernel<<<dim3(B_ * NH_, L_ / 128), 256, 0, stream>>>(q16, k16, v16, o16);
    gemm(o16, 416, wm1[t], 32,   attp[t][7], 20,   nullptr, 0, 0, h116, 32, 32, 1);
    gemm(h116, 32, wm2[t], 1008, attp[t][9], 1000, hF, 1000, (t > 0) ? 1 : 0,
         nullptr, 0, 0, 0);
    norm_kernel<<<M_, 256, 0, stream>>>(hF, na[t], nb[t], n16, 1000, 1000, 1024);
    Ablk = n16; ldA = 1024;
  }

  // final MLP: 1000 -> 1000 (leaky) -> 16
  (void)hipMemsetAsync(g16, 0, (size_t)M_ * 1024 * 2, stream); // zero Kpad cols 1008..1023
  gemm(n16, 1024, wf1, 1008, f1b, 1000, nullptr, 0, 0, g16, 1024, 1008, 1);
  gemm(g16, 1024, wf2, 16,   f2b, 16,   hh, 16, 0, nullptr, 0, 0, 0);

  ll_kernel<<<B_, 256, 0, stream>>>(hh, x, objective, (float*)d_out + 16384);
  (void)hipMemcpyAsync(d_out, x, (size_t)16384 * sizeof(float),
                       hipMemcpyDeviceToDevice, stream);
}
